// Model_44882408243652
// MI455X (gfx1250) — compile-verified, matrix operand order checked by reading, not run on hardware
//
#include <hip/hip_runtime.h>
#include <math.h>

typedef __attribute__((ext_vector_type(16))) __bf16 v16bf;
typedef __attribute__((ext_vector_type(8)))  float  v8f;

#define H       200
#define HP      208            // H padded to 16
#define KH      224            // H padded to 32 (as contraction dim)
#define IN_F    768
#define G4      800            // 4*H
#define NT_G    50             // G4/16 output tiles
#define NPAIR   25             // NT_G/2
#define KENC    992            // 768 (x) + 224 (h, padded)
#define KDEC    416            // 208 (in) + 208 (h)
#define NT_M1   13             // HP/16
#define B_TOT   256
#define T_ENC   512
#define T_DEC   326
#define BCHUNK  16
#define NCH     9
#define OUT_CH  (B_TOT*T_DEC*NCH)      // 751104, KL lives at d_out[OUT_CH]

// ---- workspace layout (bytes) ----
#define WS_WF   0
#define WS_WB   (WS_WF + G4*KENC*2)
#define WS_WC   (WS_WB + G4*KENC*2)
#define WS_W1   (WS_WC + G4*KDEC*2)
#define WS_W2   (WS_W1 + HP*KH*2)
#define WS_BSF  ((WS_W2 + 16*KH*2 + 255) & ~255)
#define WS_BSB  (WS_BSF + G4*4)
#define WS_BSC  (WS_BSB + G4*4)
#define WS_HF   (WS_BSC + G4*4)
#define WS_Z    (WS_HF + 2*B_TOT*H*4)

__device__ __forceinline__ float sigf(float x) { return 1.0f / (1.0f + __expf(-x)); }

#define WMMA_BF16(a, b, c) \
    __builtin_amdgcn_wmma_f32_16x16x32_bf16(false, (a), false, (b), (short)0, (c), false, false)

// ---------------- weight prep: f32 -> padded bf16, fused biases ----------------
__global__ __launch_bounds__(256)
void prep_kernel(const float* wif, const float* whf, const float* bif, const float* bhf,
                 const float* wib, const float* whb, const float* bib, const float* bhb,
                 const float* wic, const float* whc, const float* bic, const float* bhc,
                 const float* W1, const float* W2, char* ws, float* d_out)
{
    const int gid = blockIdx.x * blockDim.x + threadIdx.x;
    const int stride = gridDim.x * blockDim.x;
    __bf16* Wf  = (__bf16*)(ws + WS_WF);
    __bf16* Wb  = (__bf16*)(ws + WS_WB);
    __bf16* Wc  = (__bf16*)(ws + WS_WC);
    __bf16* W1p = (__bf16*)(ws + WS_W1);
    __bf16* W2p = (__bf16*)(ws + WS_W2);
    float* bsf = (float*)(ws + WS_BSF);
    float* bsb = (float*)(ws + WS_BSB);
    float* bsc = (float*)(ws + WS_BSC);

    for (int i = gid; i < G4 * KENC; i += stride) {
        int n = i / KENC, k = i % KENC;
        float vf = (k < IN_F) ? wif[n*IN_F + k] : ((k < IN_F + H) ? whf[n*H + (k - IN_F)] : 0.0f);
        float vb = (k < IN_F) ? wib[n*IN_F + k] : ((k < IN_F + H) ? whb[n*H + (k - IN_F)] : 0.0f);
        Wf[i] = (__bf16)vf;  Wb[i] = (__bf16)vb;
    }
    for (int i = gid; i < G4 * KDEC; i += stride) {
        int n = i / KDEC, k = i % KDEC;
        float v = (k < H) ? wic[n*H + k] : ((k >= HP && k < HP + H) ? whc[n*H + (k - HP)] : 0.0f);
        Wc[i] = (__bf16)v;
    }
    for (int i = gid; i < HP * KH; i += stride) {
        int n = i / KH, k = i % KH;
        W1p[i] = (__bf16)((n < H && k < H) ? W1[n*H + k] : 0.0f);
    }
    for (int i = gid; i < 16 * KH; i += stride) {
        int n = i / KH, k = i % KH;
        W2p[i] = (__bf16)((n < NCH && k < H) ? W2[n*H + k] : 0.0f);
    }
    for (int i = gid; i < G4; i += stride) {
        bsf[i] = bif[i] + bhf[i];
        bsb[i] = bib[i] + bhb[i];
        bsc[i] = bic[i] + bhc[i];
    }
    if (gid == 0) d_out[OUT_CH] = 0.0f;
}

// ---------------- fused encoder bi-LSTM scan (one WG = one dir x 16 batch rows) --------
__global__ __launch_bounds__(256)
void encoder_kernel(const float* __restrict__ input, char* __restrict__ ws)
{
    const int dir   = blockIdx.x >> 4;
    const int b0    = (blockIdx.x & 15) * BCHUNK;
    extern __shared__ __align__(32) char smem[];
    __bf16* Abuf = (__bf16*)smem;                               // [16][992] bf16
    float*  gbuf = (float*)(smem + BCHUNK*KENC*2);              // [16][800]
    float*  cbuf = gbuf + BCHUNK*G4;                            // [16][200]
    float*  hbuf = cbuf + BCHUNK*H;                             // [16][200]
    float*  bsumL= hbuf + BCHUNK*H;                             // [800]

    const __bf16* W    = (const __bf16*)(ws + (dir ? WS_WB : WS_WF));
    const float*  bsum = (const float*)(ws + (dir ? WS_BSB : WS_BSF));
    const int tid = threadIdx.x;

    // init: h0 = c0 = 0, zero the padded h-section of A, cache fused biases
    for (int i = tid; i < BCHUNK*KH; i += 256) {
        int r = i / KH, c = i % KH;
        Abuf[r*KENC + IN_F + c] = (__bf16)0.0f;
    }
    for (int i = tid; i < BCHUNK*H; i += 256) { cbuf[i] = 0.0f; hbuf[i] = 0.0f; }
    for (int i = tid; i < G4; i += 256) bsumL[i] = bsum[i];
    __syncthreads();

    const int wave = tid >> 5, lane = tid & 31;
    const int rsel = lane & 15, kseg = (lane >> 4) * 16;
    const int rbase = (lane >> 4) * 8;
    const __bf16* arow = Abuf + rsel*KENC + kseg;

    for (int t = 0; t < T_ENC; ++t) {
        const int tt = dir ? (T_ENC - 1 - t) : t;
        // stage x_t (f32 -> bf16) into A[:, 0:768]
        for (int i = tid; i < BCHUNK*IN_F; i += 256) {
            int r = i / IN_F, c = i % IN_F;
            Abuf[r*KENC + c] =
                (__bf16)input[(size_t)(b0 + r)*T_ENC*IN_F + (size_t)tt*IN_F + c];
        }
        // prefetch x_{t+1} cache lines (overlaps with the WMMA phase below)
        if (t + 1 < T_ENC) {
            const int tn = dir ? (T_ENC - 2 - t) : (t + 1);
            for (int i = tid; i < BCHUNK * 24; i += 256) {
                int r = i / 24, c = (i % 24) * 32;   // one probe per 128B
                __builtin_prefetch(
                    &input[(size_t)(b0 + r)*T_ENC*IN_F + (size_t)tn*IN_F + c], 0, 0);
            }
        }
        __syncthreads();
        // g = [x|h] @ W^T — pairs of N-tiles, software-pipelined B fragments
        for (int p = wave; p < NPAIR; p += 8) {
            const __bf16* wrow0 = W + (size_t)(2*p*16 + rsel)*KENC + kseg;
            const __bf16* wrow1 = wrow0 + (size_t)16*KENC;
            v8f acc0 = {}, acc1 = {};
            v16bf bf0 = *(const v16bf*)(wrow0);
            v16bf bf1 = *(const v16bf*)(wrow1);
            int k = 0;
            for (; k < KENC - 32; k += 32) {
                v16bf bn0 = *(const v16bf*)(wrow0 + k + 32);
                v16bf bn1 = *(const v16bf*)(wrow1 + k + 32);
                v16bf a   = *(const v16bf*)(arow + k);
                acc0 = WMMA_BF16(a, bf0, acc0);
                acc1 = WMMA_BF16(a, bf1, acc1);
                bf0 = bn0;  bf1 = bn1;
            }
            {
                v16bf a = *(const v16bf*)(arow + k);
                acc0 = WMMA_BF16(a, bf0, acc0);
                acc1 = WMMA_BF16(a, bf1, acc1);
            }
            const int col0 = 2*p*16 + rsel;
            #pragma unroll
            for (int r = 0; r < 8; ++r) {
                gbuf[(rbase + r)*G4 + col0]      = acc0[r];
                gbuf[(rbase + r)*G4 + col0 + 16] = acc1[r];
            }
        }
        __syncthreads();
        // LSTM pointwise update (f32), write new h back as bf16 into A[:, 768:]
        for (int i = tid; i < BCHUNK*H; i += 256) {
            int r = i / H, j = i % H;
            float gi = gbuf[r*G4 +       j] + bsumL[      j];
            float gf = gbuf[r*G4 + H   + j] + bsumL[H   + j];
            float gg = gbuf[r*G4 + 2*H + j] + bsumL[2*H + j];
            float go = gbuf[r*G4 + 3*H + j] + bsumL[3*H + j];
            float c  = sigf(gf)*cbuf[i] + sigf(gi)*tanhf(gg);
            float h  = sigf(go)*tanhf(c);
            cbuf[i] = c;  hbuf[i] = h;
            Abuf[r*KENC + IN_F + j] = (__bf16)h;
        }
        __syncthreads();
    }
    float* hf = (float*)(ws + WS_HF) + (size_t)dir*B_TOT*H;
    for (int i = tid; i < BCHUNK*H; i += 256) {
        int r = i / H, j = i % H;
        hf[(b0 + r)*H + j] = hbuf[i];
    }
}

// ---------------- VAE heads + reparam + KL (tiny, plain f32) ----------------
__global__ __launch_bounds__(256)
void vae_kernel(const float* eps, const float* W_mu, const float* b_mu,
                const float* W_lv, const float* b_lv, char* ws, float* d_out)
{
    const int b = blockIdx.x;
    __shared__ float temp[H];
    __shared__ float red[256];
    const float* hf0 = (const float*)(ws + WS_HF);
    const float* hf1 = hf0 + B_TOT*H;
    float* zbuf = (float*)(ws + WS_Z);
    const int tid = threadIdx.x;

    for (int j = tid; j < H; j += 256) temp[j] = hf0[b*H + j] + hf1[b*H + j];
    __syncthreads();

    float kl = 0.0f;
    for (int j = tid; j < H; j += 256) {
        float mu = b_mu[j], lv = b_lv[j];
        for (int k = 0; k < H; ++k) {
            mu += W_mu[j*H + k] * temp[k];
            lv += W_lv[j*H + k] * temp[k];
        }
        zbuf[b*H + j] = mu + eps[b*H + j] * __expf(0.5f * lv);
        kl += mu*mu + __expf(lv) - lv - 1.0f;
    }
    red[tid] = kl;  __syncthreads();
    for (int s = 128; s > 0; s >>= 1) {
        if (tid < s) red[tid] += red[tid + s];
        __syncthreads();
    }
    if (tid == 0)
        atomicAdd(&d_out[OUT_CH], red[0] * (0.5f * 0.01f / (float)(B_TOT * H)));
}

// ---------------- decoder scan: LSTMCell + chord MLP, all WMMA ----------------
__global__ __launch_bounds__(256)
void decoder_kernel(const float* hx0, const float* cx0,
                    const float* b1, const float* b2,
                    char* __restrict__ ws, float* __restrict__ d_out)
{
    const int b0 = blockIdx.x * BCHUNK;
    extern __shared__ __align__(32) char smem[];
    __bf16* Abuf = (__bf16*)smem;                     // [16][416]
    __bf16* A2   = Abuf + BCHUNK*KDEC;                // [16][224] h for MLP
    __bf16* Rbuf = A2   + BCHUNK*KH;                  // [16][224] relu(h@W1+b1)
    float*  gbuf = (float*)(Rbuf + BCHUNK*KH);        // [16][800]
    float*  cbuf = gbuf + BCHUNK*G4;                  // [16][200]
    float*  bsumL= cbuf + BCHUNK*H;                   // [800]
    float*  b1L  = bsumL + G4;                        // [200]
    float*  b2L  = b1L + H;                           // [16]

    const __bf16* Wc  = (const __bf16*)(ws + WS_WC);
    const __bf16* W1p = (const __bf16*)(ws + WS_W1);
    const __bf16* W2p = (const __bf16*)(ws + WS_W2);
    const float*  bsc = (const float*)(ws + WS_BSC);
    const float*  zbuf= (const float*)(ws + WS_Z);
    const int tid = threadIdx.x;

    // init state: A = [z | hx0], c = cx0, zero pads, cache biases
    for (int i = tid; i < BCHUNK*KH; i += 256) { A2[i] = (__bf16)0.0f; Rbuf[i] = (__bf16)0.0f; }
    for (int i = tid; i < BCHUNK*8; i += 256) {
        int r = i / 8, c = i % 8;
        Abuf[r*KDEC + H + c]      = (__bf16)0.0f;     // pad 200..207
        Abuf[r*KDEC + HP + H + c] = (__bf16)0.0f;     // pad 408..415
    }
    for (int i = tid; i < BCHUNK*H; i += 256) {
        int r = i / H, j = i % H;
        float z  = zbuf[(b0 + r)*H + j];
        float h0 = hx0 [(b0 + r)*H + j];
        Abuf[r*KDEC + j]      = (__bf16)z;
        Abuf[r*KDEC + HP + j] = (__bf16)h0;
        A2  [r*KH + j]        = (__bf16)h0;
        cbuf[i] = cx0[(b0 + r)*H + j];
    }
    for (int i = tid; i < G4; i += 256) bsumL[i] = bsc[i];
    for (int i = tid; i < H;  i += 256) b1L[i] = b1[i];
    for (int i = tid; i < 16; i += 256) b2L[i] = (i < NCH) ? b2[i] : 0.0f;
    __syncthreads();

    const int wave = tid >> 5, lane = tid & 31;
    const int rsel = lane & 15, kseg = (lane >> 4) * 16;
    const int rbase = (lane >> 4) * 8;
    const __bf16* arow  = Abuf + rsel*KDEC + kseg;
    const __bf16* arow2 = A2   + rsel*KH   + kseg;
    const __bf16* arow3 = Rbuf + rsel*KH   + kseg;

    for (int t = 0; t < T_DEC; ++t) {
        // gates: [x|h] @ Wc^T — paired N-tiles, pipelined B fragments
        for (int p = wave; p < NPAIR; p += 8) {
            const __bf16* wrow0 = Wc + (size_t)(2*p*16 + rsel)*KDEC + kseg;
            const __bf16* wrow1 = wrow0 + (size_t)16*KDEC;
            v8f acc0 = {}, acc1 = {};
            v16bf bf0 = *(const v16bf*)(wrow0);
            v16bf bf1 = *(const v16bf*)(wrow1);
            int k = 0;
            for (; k < KDEC - 32; k += 32) {
                v16bf bn0 = *(const v16bf*)(wrow0 + k + 32);
                v16bf bn1 = *(const v16bf*)(wrow1 + k + 32);
                v16bf a   = *(const v16bf*)(arow + k);
                acc0 = WMMA_BF16(a, bf0, acc0);
                acc1 = WMMA_BF16(a, bf1, acc1);
                bf0 = bn0;  bf1 = bn1;
            }
            {
                v16bf a = *(const v16bf*)(arow + k);
                acc0 = WMMA_BF16(a, bf0, acc0);
                acc1 = WMMA_BF16(a, bf1, acc1);
            }
            const int col0 = 2*p*16 + rsel;
            #pragma unroll
            for (int r = 0; r < 8; ++r) {
                gbuf[(rbase + r)*G4 + col0]      = acc0[r];
                gbuf[(rbase + r)*G4 + col0 + 16] = acc1[r];
            }
        }
        __syncthreads();
        // pointwise cell update; next-step input is h itself (write both A sections)
        for (int i = tid; i < BCHUNK*H; i += 256) {
            int r = i / H, j = i % H;
            float gi = gbuf[r*G4 +       j] + bsumL[      j];
            float gf = gbuf[r*G4 + H   + j] + bsumL[H   + j];
            float gg = gbuf[r*G4 + 2*H + j] + bsumL[2*H + j];
            float go = gbuf[r*G4 + 3*H + j] + bsumL[3*H + j];
            float c  = sigf(gf)*cbuf[i] + sigf(gi)*tanhf(gg);
            float h  = sigf(go)*tanhf(c);
            cbuf[i] = c;
            __bf16 hb = (__bf16)h;
            Abuf[r*KDEC + j]      = hb;
            Abuf[r*KDEC + HP + j] = hb;
            A2  [r*KH + j]        = hb;
        }
        __syncthreads();
        // MLP layer 1: relu(h @ W1^T + b1) — pipelined B
        for (int n = wave; n < NT_M1; n += 8) {
            const __bf16* wrow = W1p + (size_t)(n*16 + rsel)*KH + kseg;
            v8f acc = {};
            v16bf bf = *(const v16bf*)(wrow);
            int k = 0;
            for (; k < KH - 32; k += 32) {
                v16bf bn = *(const v16bf*)(wrow + k + 32);
                v16bf a  = *(const v16bf*)(arow2 + k);
                acc = WMMA_BF16(a, bf, acc);
                bf = bn;
            }
            {
                v16bf a = *(const v16bf*)(arow2 + k);
                acc = WMMA_BF16(a, bf, acc);
            }
            const int col = n*16 + rsel;
            #pragma unroll
            for (int r = 0; r < 8; ++r) {
                float v = (col < H) ? fmaxf(acc[r] + b1L[col], 0.0f) : 0.0f;
                Rbuf[(rbase + r)*KH + col] = (__bf16)v;
            }
        }
        __syncthreads();
        // MLP layer 2: r @ W2^T + b2 -> chords (one 16x16 tile; wave 0)
        if (wave == 0) {
            v8f acc = {};
            const __bf16* wrow = W2p + (size_t)rsel*KH + kseg;
            #pragma unroll
            for (int k = 0; k < KH; k += 32) {
                v16bf a = *(const v16bf*)(arow3 + k);
                v16bf b = *(const v16bf*)(wrow + k);
                acc = WMMA_BF16(a, b, acc);
            }
            const int col = rsel;
            if (col < NCH) {
                #pragma unroll
                for (int r = 0; r < 8; ++r) {
                    int b = b0 + rbase + r;
                    d_out[(size_t)b*T_DEC*NCH + (size_t)t*NCH + col] = acc[r] + b2L[col];
                }
            }
        }
        __syncthreads();
    }
}

// ---------------- launch ----------------
extern "C" void kernel_launch(void* const* d_in, const int* in_sizes, int n_in,
                              void* d_out, int out_size, void* d_ws, size_t ws_size,
                              hipStream_t stream)
{
    const float* input = (const float*)d_in[0];
    const float* eps   = (const float*)d_in[1];
    const float* hx0   = (const float*)d_in[2];
    const float* cx0   = (const float*)d_in[3];
    const float* wif = (const float*)d_in[4],  *whf = (const float*)d_in[5];
    const float* bif = (const float*)d_in[6],  *bhf = (const float*)d_in[7];
    const float* wib = (const float*)d_in[8],  *whb = (const float*)d_in[9];
    const float* bib = (const float*)d_in[10], *bhb = (const float*)d_in[11];
    const float* W_mu = (const float*)d_in[12], *b_mu = (const float*)d_in[13];
    const float* W_lv = (const float*)d_in[14], *b_lv = (const float*)d_in[15];
    const float* wic = (const float*)d_in[16], *whc = (const float*)d_in[17];
    const float* bic = (const float*)d_in[18], *bhc = (const float*)d_in[19];
    const float* W1  = (const float*)d_in[20], *b1  = (const float*)d_in[21];
    const float* W2  = (const float*)d_in[22], *b2  = (const float*)d_in[23];
    char*  ws  = (char*)d_ws;
    float* out = (float*)d_out;
    (void)in_sizes; (void)n_in; (void)out_size; (void)ws_size;

    const size_t smem_enc = (size_t)BCHUNK*KENC*2
                          + (size_t)(BCHUNK*G4 + 2*BCHUNK*H + G4)*4;
    const size_t smem_dec = (size_t)BCHUNK*KDEC*2 + 2*(size_t)BCHUNK*KH*2
                          + (size_t)(BCHUNK*G4 + BCHUNK*H + G4 + H + 16)*4;

    prep_kernel<<<256, 256, 0, stream>>>(wif, whf, bif, bhf,
                                         wib, whb, bib, bhb,
                                         wic, whc, bic, bhc,
                                         W1, W2, ws, out);
    encoder_kernel<<<32, 256, smem_enc, stream>>>(input, ws);
    vae_kernel<<<B_TOT, 256, 0, stream>>>(eps, W_mu, b_mu, W_lv, b_lv, ws, out);
    decoder_kernel<<<16, 256, smem_dec, stream>>>(hx0, cx0, b1, b2, ws, out);
}
